// xLSTM_32220844654871
// MI455X (gfx1250) — compile-verified, matrix operand order checked by reading, not run on hardware
//
#include <hip/hip_runtime.h>
#include <math.h>

// ---------------- model dims ----------------
#define B_      512
#define S_      48
#define D_      256
#define NH_     4
#define DH_     64
#define INNER_  512
#define MDH_    128
#define FFN_    384
#define ROWS_   (B_ * S_)          // 24576
#define EPS_    1e-5f

// ---------------- WMMA types ----------------
typedef __bf16 v16bf __attribute__((ext_vector_type(16)));
typedef float  v8f   __attribute__((ext_vector_type(8)));

__device__ __forceinline__ void cvt8(v16bf& d, int o, float4 x, float4 y) {
  d[o + 0] = (__bf16)x.x; d[o + 1] = (__bf16)x.y;
  d[o + 2] = (__bf16)x.z; d[o + 3] = (__bf16)x.w;
  d[o + 4] = (__bf16)y.x; d[o + 5] = (__bf16)y.y;
  d[o + 6] = (__bf16)y.z; d[o + 7] = (__bf16)y.w;
}
__device__ __forceinline__ void cvt8s(v16bf& d, int o, float4 x, float4 y, float s) {
  d[o + 0] = (__bf16)(x.x * s); d[o + 1] = (__bf16)(x.y * s);
  d[o + 2] = (__bf16)(x.z * s); d[o + 3] = (__bf16)(x.w * s);
  d[o + 4] = (__bf16)(y.x * s); d[o + 5] = (__bf16)(y.y * s);
  d[o + 6] = (__bf16)(y.z * s); d[o + 7] = (__bf16)(y.w * s);
}

__device__ __forceinline__ float wave_sum(float v) {
  #pragma unroll
  for (int m = 16; m >= 1; m >>= 1) v += __shfl_xor(v, m, 32);
  return v;
}

__device__ __forceinline__ float sigmoidf_(float x) { return 1.f / (1.f + __expf(-x)); }
__device__ __forceinline__ float siluf_(float x)    { return x * sigmoidf_(x); }
__device__ __forceinline__ float logsigf_(float x)  {
  return fminf(x, 0.f) - log1pf(__expf(-fabsf(x)));   // -log(1+exp(-x)), stable
}

// ============================================================
// Block GEMM: C[M,N] = A[M,K] @ W (+ RES). bf16 WMMA, f32 acc.
// Block tile: 128(M) x 64(N); 8 waves, each wave 16x64 (4 accs).
// W tile (32K x 64N) staged in LDS, stored [col][k] with pad 36
// so B fragments are aligned ds_load_b128s (conflict-free).
// TRANSW==0: W is [K,N]; TRANSW==1: W is [N,K].
// Requires M%128==0, N%64==0, K%32==0.
// ============================================================
template<int TRANSW, int HASRES>
__global__ __launch_bounds__(256) void gemm_wmma_kernel(
    const float* __restrict__ A, const float* __restrict__ W,
    const float* __restrict__ RES, float* __restrict__ C,
    int M, int N, int Kd, int ldA, int ldW, int ldC)
{
  __shared__ float Ws[64 * 36];           // [col][k], padded
  int tilesN = N >> 6;
  int bM = blockIdx.x / tilesN;
  int bN = blockIdx.x % tilesN;
  int tid = threadIdx.x, wv = tid >> 5, lane = tid & 31;
  int l15 = lane & 15, grp = lane >> 4;
  int row   = bM * 128 + wv * 16 + l15;
  int ncol0 = bN * 64;
  const float* arow = A + (size_t)row * ldA;

  v8f acc[4] = {{}, {}, {}, {}};
  for (int k0 = 0; k0 < Kd; k0 += 32) {
    __syncthreads();
    if (TRANSW) {
      // W[N][K]: coalesced along K
      for (int i = tid; i < 2048; i += 256) {
        int c = i >> 5, kk = i & 31;
        Ws[c * 36 + kk] = W[(size_t)(ncol0 + c) * ldW + k0 + kk];
      }
    } else {
      // W[K][N]: coalesced along N
      for (int i = tid; i < 2048; i += 256) {
        int kk = i >> 6, c = i & 63;
        Ws[c * 36 + kk] = W[(size_t)(k0 + kk) * ldW + ncol0 + c];
      }
    }
    __syncthreads();

    __builtin_prefetch(arow + k0 + 32, 0, 1);   // speculative; dropped if invalid
    v16bf a;
    cvt8(a, 0, *(const float4*)(arow + k0 + grp * 8),
               *(const float4*)(arow + k0 + grp * 8 + 4));
    cvt8(a, 8, *(const float4*)(arow + k0 + 16 + grp * 8),
               *(const float4*)(arow + k0 + 16 + grp * 8 + 4));

    #pragma unroll
    for (int nt = 0; nt < 4; ++nt) {
      const float* wc = &Ws[(nt * 16 + l15) * 36];
      v16bf b;
      cvt8(b, 0, *(const float4*)(wc + grp * 8),
                 *(const float4*)(wc + grp * 8 + 4));
      cvt8(b, 8, *(const float4*)(wc + 16 + grp * 8),
                 *(const float4*)(wc + 16 + grp * 8 + 4));
      acc[nt] = __builtin_amdgcn_wmma_f32_16x16x32_bf16(false, a, false, b,
                                                        (short)0, acc[nt], false, false);
    }
  }

  #pragma unroll
  for (int nt = 0; nt < 4; ++nt) {
    #pragma unroll
    for (int r = 0; r < 8; ++r) {
      int m = bM * 128 + wv * 16 + r + grp * 8;
      int c = ncol0 + nt * 16 + l15;
      float v = acc[nt][r];
      if (HASRES) v += RES[(size_t)m * ldC + c];
      C[(size_t)m * ldC + c] = v;
    }
  }
}

// ============================================================
// LayerNorm (no bias): one wave per row
// ============================================================
__global__ void ln_kernel(const float* __restrict__ X, const float* __restrict__ Wt,
                          float* __restrict__ Y, int rows, int L)
{
  int wid = blockIdx.x * (blockDim.x >> 5) + (threadIdx.x >> 5);
  int lane = threadIdx.x & 31;
  if (wid >= rows) return;
  const float* xr = X + (size_t)wid * L;
  float s = 0.f, s2 = 0.f;
  for (int i = lane; i < L; i += 32) { float v = xr[i]; s += v; s2 += v * v; }
  s = wave_sum(s); s2 = wave_sum(s2);
  float mean = s / L, var = s2 / L - mean * mean;
  float inv = rsqrtf(var + EPS_);
  float* yr = Y + (size_t)wid * L;
  for (int i = lane; i < L; i += 32) yr[i] = (xr[i] - mean) * inv * Wt[i];
}

// ============================================================
// Causal conv (K=4) + SiLU, channelwise
// ============================================================
__global__ void conv_silu_kernel(const float* __restrict__ X, const float* __restrict__ CW,
                                 const float* __restrict__ CB, float* __restrict__ Y,
                                 int Ch, int ldX)
{
  int n = blockIdx.x * blockDim.x + threadIdx.x;
  if (n >= ROWS_ * Ch) return;
  int c = n % Ch, row = n / Ch, s = row % S_;
  float acc = CB[c];
  #pragma unroll
  for (int k = 0; k < 4; ++k) {
    int sp = s + k - 3;
    if (sp >= 0) acc += X[(size_t)(row + k - 3) * ldX + c] * CW[k * Ch + c];
  }
  Y[(size_t)row * Ch + c] = siluf_(acc);
}

// ============================================================
// Headwise 4x4 per-head matmul (q/k/v of mLSTM): pure VALU
// ============================================================
__global__ void headwise4_kernel(const float* __restrict__ X, const float* __restrict__ Wt,
                                 float* __restrict__ Y, int ldX)
{
  int n = blockIdx.x * blockDim.x + threadIdx.x;
  if (n >= ROWS_ * INNER_) return;
  int d = n & 511, row = n >> 9;
  int h = d >> 2, oo = d & 3;
  const float* xr = X + (size_t)row * ldX + (h << 2);
  const float* wr = Wt + h * 16 + oo * 4;
  Y[(size_t)row * INNER_ + d] =
      xr[0] * wr[0] + xr[1] * wr[1] + xr[2] * wr[2] + xr[3] * wr[3];
}

// ============================================================
// ig/fg gates: one wave per (b,s); outputs transposed to (B,NH,S)
// ============================================================
__global__ void gate_kernel(const float* __restrict__ Q, const float* __restrict__ Kb,
                            const float* __restrict__ V, const float* __restrict__ IGW,
                            const float* __restrict__ IGB, const float* __restrict__ FGW,
                            const float* __restrict__ FGB,
                            float* __restrict__ IG, float* __restrict__ FG)
{
  int wid = blockIdx.x * (blockDim.x >> 5) + (threadIdx.x >> 5);
  int lane = threadIdx.x & 31;
  if (wid >= ROWS_) return;
  int b = wid / S_, s = wid % S_;
  float aI[4] = {0,0,0,0}, aF[4] = {0,0,0,0};
  size_t base = (size_t)wid * INNER_;
  for (int j = lane; j < INNER_; j += 32) {
    float qv = Q[base + j], kv = Kb[base + j], vv = V[base + j];
    const float* wiq = IGW + (size_t)j * 4;
    const float* wik = IGW + (size_t)(INNER_ + j) * 4;
    const float* wiv = IGW + (size_t)(2 * INNER_ + j) * 4;
    const float* wfq = FGW + (size_t)j * 4;
    const float* wfk = FGW + (size_t)(INNER_ + j) * 4;
    const float* wfv = FGW + (size_t)(2 * INNER_ + j) * 4;
    #pragma unroll
    for (int g = 0; g < 4; ++g) {
      aI[g] += qv * wiq[g] + kv * wik[g] + vv * wiv[g];
      aF[g] += qv * wfq[g] + kv * wfk[g] + vv * wfv[g];
    }
  }
  #pragma unroll
  for (int g = 0; g < 4; ++g) { aI[g] = wave_sum(aI[g]); aF[g] = wave_sum(aF[g]); }
  if (lane == 0) {
    #pragma unroll
    for (int g = 0; g < 4; ++g) {
      IG[((size_t)b * NH_ + g) * S_ + s] = aI[g] + IGB[g];
      FG[((size_t)b * NH_ + g) * S_ + s] = aF[g] + FGB[g];
    }
  }
}

// ============================================================
// mLSTM parallel attention per (b,h), fully fused
// ============================================================
__global__ __launch_bounds__(256) void attn_kernel(
    const float* __restrict__ Q, const float* __restrict__ Kb, const float* __restrict__ V,
    const float* __restrict__ IG, const float* __restrict__ FG,
    const float* __restrict__ XC, const float* __restrict__ UP,
    const float* __restrict__ ONW, const float* __restrict__ SKIP,
    float* __restrict__ H)
{
  __shared__ float lc[S_], igv[S_], maxD[S_], invn[S_];
  __shared__ float Cm[S_ * 64];
  __shared__ float Hm[S_ * MDH_];

  int bh = blockIdx.x, b = bh / NH_, h = bh % NH_;
  int tid = threadIdx.x, wv = tid >> 5, lane = tid & 31;
  int grp = lane >> 4, l15 = lane & 15;
  size_t hoff = (size_t)b * S_ * INNER_ + (size_t)h * MDH_;

  for (int i = tid; i < S_ * 64; i += 256) Cm[i] = 0.f;
  if (tid == 0) {
    float run = 0.f, pm = -1e30f;
    for (int s = 0; s < S_; ++s) {
      float f = FG[((size_t)b * NH_ + h) * S_ + s];
      run += logsigf_(f);
      lc[s] = run;
      float iv = IG[((size_t)b * NH_ + h) * S_ + s];
      igv[s] = iv;
      pm = fmaxf(pm, iv - run);
      maxD[s] = run + pm;
    }
  }
  __syncthreads();

  const float scale = 0.08838834764831845f;  // 1/sqrt(MDH)
  // ---- phase A: C = (q @ k^T) * Dm / sqrt(MDH), 3x3 tiles ----
  for (int tt = wv; tt < 9; tt += 8) {
    int ti = tt / 3, tj = tt % 3;
    int srow = ti * 16 + l15, tcol = tj * 16 + l15;
    const float* qrow = Q  + hoff + (size_t)srow * INNER_;
    const float* krow = Kb + hoff + (size_t)tcol * INNER_;
    v8f acc = {};
    for (int k0 = 0; k0 < MDH_; k0 += 32) {
      v16bf a, bm;
      cvt8(a, 0, *(const float4*)(qrow + k0 + grp * 8),
                 *(const float4*)(qrow + k0 + grp * 8 + 4));
      cvt8(a, 8, *(const float4*)(qrow + k0 + 16 + grp * 8),
                 *(const float4*)(qrow + k0 + 16 + grp * 8 + 4));
      cvt8(bm, 0, *(const float4*)(krow + k0 + grp * 8),
                  *(const float4*)(krow + k0 + grp * 8 + 4));
      cvt8(bm, 8, *(const float4*)(krow + k0 + 16 + grp * 8),
                  *(const float4*)(krow + k0 + 16 + grp * 8 + 4));
      acc = __builtin_amdgcn_wmma_f32_16x16x32_bf16(false, a, false, bm,
                                                    (short)0, acc, false, false);
    }
    float lct = lc[tcol], igt = igv[tcol];
    #pragma unroll
    for (int r = 0; r < 8; ++r) {
      int s = ti * 16 + r + grp * 8;
      // branchless causal mask: exp(-1e30) == 0 for masked entries
      float dmask = (tcol <= s) ? 0.f : -1e30f;
      float logd  = lc[s] - lct + igt - maxD[s] + dmask;
      Cm[s * 64 + tcol] = acc[r] * scale * __expf(logd);
    }
  }
  __syncthreads();

  if (tid < S_) {
    float sm = 0.f;
    for (int t = 0; t < 64; ++t) sm += Cm[tid * 64 + t];
    float nr = fmaxf(fabsf(sm), __expf(-maxD[tid]));
    invn[tid] = 1.f / (nr + 1e-6f);
  }
  __syncthreads();

  // ---- phase B: Hm = (C/norm) @ V, 3x8 tiles, K=64 (zero-padded) ----
  for (int tt = wv; tt < 24; tt += 8) {
    int ti = tt / 8, tn = tt % 8;
    int srow = ti * 16 + l15, dcol = tn * 16 + l15;
    float isc = invn[srow];
    const float* crow = &Cm[srow * 64];
    v8f acc = {};
    for (int k0 = 0; k0 < 64; k0 += 32) {
      v16bf a, bm;
      cvt8s(a, 0, *(const float4*)(crow + k0 + grp * 8),
                  *(const float4*)(crow + k0 + grp * 8 + 4), isc);
      cvt8s(a, 8, *(const float4*)(crow + k0 + 16 + grp * 8),
                  *(const float4*)(crow + k0 + 16 + grp * 8 + 4), isc);
      #pragma unroll
      for (int e = 0; e < 16; ++e) {
        int kk = (e < 8) ? (k0 + grp * 8 + e) : (k0 + 16 + grp * 8 + (e - 8));
        bm[e] = (kk < S_) ? (__bf16)V[hoff + (size_t)kk * INNER_ + dcol] : (__bf16)0.f;
      }
      acc = __builtin_amdgcn_wmma_f32_16x16x32_bf16(false, a, false, bm,
                                                    (short)0, acc, false, false);
    }
    #pragma unroll
    for (int r = 0; r < 8; ++r) {
      int s = ti * 16 + r + grp * 8;
      Hm[s * MDH_ + dcol] = acc[r];
    }
  }
  __syncthreads();

  // ---- phase C: mh_norm + gating ----
  for (int s = wv; s < S_; s += 8) {
    float sm = 0.f, s2 = 0.f;
    for (int i = lane; i < MDH_; i += 32) { float v = Hm[s * MDH_ + i]; sm += v; s2 += v * v; }
    sm = wave_sum(sm); s2 = wave_sum(s2);
    float mean = sm / MDH_, var = s2 / MDH_ - mean * mean;
    float inv = rsqrtf(var + EPS_);
    size_t rbase = ((size_t)b * S_ + s) * INNER_ + (size_t)h * MDH_;
    size_t zbase = ((size_t)b * S_ + s) * 1024 + INNER_ + (size_t)h * MDH_;
    for (int i = lane; i < MDH_; i += 32) {
      float hn  = (Hm[s * MDH_ + i] - mean) * inv * ONW[h * MDH_ + i];
      float xcv = XC[rbase + i];
      float zv  = UP[zbase + i];
      H[rbase + i] = (hn + SKIP[h * MDH_ + i] * xcv) * siluf_(zv);
    }
  }
}

// ============================================================
// sLSTM pointwise recurrence step
// ============================================================
__global__ void slstm_step_kernel(
    const float* __restrict__ RAW,
    const float* __restrict__ GI, const float* __restrict__ GF,
    const float* __restrict__ GZ, const float* __restrict__ GO,
    const float* __restrict__ BIAS,
    float* __restrict__ Hs, float* __restrict__ Cs,
    float* __restrict__ Ns, float* __restrict__ Ms,
    float* __restrict__ YS, int t)
{
  int n = blockIdx.x * blockDim.x + threadIdx.x;
  if (n >= B_ * NH_ * DH_) return;
  int b = n >> 8, hd = n & 255, h = hd >> 6, k = hd & 63;
  size_t rb = (size_t)b * 1024 + (size_t)h * 256;
  size_t gb = ((size_t)b * S_ + t) * D_ + hd;
  float iraw = RAW[rb +   0 + k] + GI[gb] + BIAS[      h * 64 + k];
  float fraw = RAW[rb +  64 + k] + GF[gb] + BIAS[256 + h * 64 + k];
  float zraw = RAW[rb + 128 + k] + GZ[gb] + BIAS[512 + h * 64 + k];
  float oraw = RAW[rb + 192 + k] + GO[gb] + BIAS[768 + h * 64 + k];
  float lfp = Ms[n] + logsigf_(fraw);
  float mn  = fmaxf(iraw, lfp);
  float ig  = __expf(iraw - mn), fg = __expf(lfp - mn);
  float cn  = fg * Cs[n] + ig * tanhf(zraw);
  float nn  = fg * Ns[n] + ig;
  float hn  = sigmoidf_(oraw) * cn / nn;
  Hs[n] = hn; Cs[n] = cn; Ns[n] = nn; Ms[n] = mn;
  YS[gb] = hn;
}

// ============================================================
// GroupNorm(DH) over (b,s,h) + residual
// ============================================================
__global__ void gn_res_kernel(const float* __restrict__ YS, const float* __restrict__ GW,
                              const float* __restrict__ XOLD, float* __restrict__ XNEW)
{
  int wid = blockIdx.x * (blockDim.x >> 5) + (threadIdx.x >> 5);
  int lane = threadIdx.x & 31;
  if (wid >= B_ * S_ * NH_) return;
  int h = wid % NH_, rs = wid / NH_;
  size_t base = (size_t)rs * D_ + (size_t)h * DH_;
  float s = 0.f, s2 = 0.f;
  for (int i = lane; i < DH_; i += 32) { float v = YS[base + i]; s += v; s2 += v * v; }
  s = wave_sum(s); s2 = wave_sum(s2);
  float mean = s / DH_, var = s2 / DH_ - mean * mean;
  float inv = rsqrtf(var + EPS_);
  for (int i = lane; i < DH_; i += 32)
    XNEW[base + i] = XOLD[base + i] + (YS[base + i] - mean) * inv * GW[h * DH_ + i];
}

// ============================================================
// GEGLU: mid = gelu_exact(gate) * up
// ============================================================
__global__ void geglu_kernel(const float* __restrict__ UP2, float* __restrict__ MID)
{
  int n = blockIdx.x * blockDim.x + threadIdx.x;
  if (n >= ROWS_ * FFN_) return;
  int j = n % FFN_, row = n / FFN_;
  float g = UP2[(size_t)row * (2 * FFN_) + j];
  float u = UP2[(size_t)row * (2 * FFN_) + FFN_ + j];
  float ge = 0.5f * g * (1.f + erff(g * 0.70710678118654752f));
  MID[(size_t)row * FFN_ + j] = ge * u;
}

// ============================================================
// Fused post-LN + MLP head on last 24 tokens
// ============================================================
__global__ void head_kernel(const float* __restrict__ X, const float* __restrict__ PW,
                            const float* __restrict__ W1, const float* __restrict__ B1,
                            const float* __restrict__ W2, const float* __restrict__ B2,
                            const float* __restrict__ WO, const float* __restrict__ BO,
                            float* __restrict__ OUT)
{
  int n = blockIdx.x * blockDim.x + threadIdx.x;
  if (n >= B_ * 24) return;
  int b = n / 24, sp = n % 24, s = S_ - 24 + sp;
  const float* xr = X + ((size_t)b * S_ + s) * D_;
  float sm = 0.f, s2 = 0.f;
  for (int i = 0; i < D_; ++i) { float v = xr[i]; sm += v; s2 += v * v; }
  float mean = sm / D_, var = s2 / D_ - mean * mean;
  float inv = rsqrtf(var + EPS_);
  float a1[10];
  #pragma unroll
  for (int j = 0; j < 10; ++j) a1[j] = B1[j];
  for (int i = 0; i < D_; ++i) {
    float xn = (xr[i] - mean) * inv * PW[i];
    #pragma unroll
    for (int j = 0; j < 10; ++j) a1[j] += xn * W1[i * 10 + j];
  }
  #pragma unroll
  for (int j = 0; j < 10; ++j) a1[j] = fmaxf(a1[j], 0.f);
  float a2[10];
  #pragma unroll
  for (int j = 0; j < 10; ++j) {
    float acc = B2[j];
    #pragma unroll
    for (int i = 0; i < 10; ++i) acc += a1[i] * W2[i * 10 + j];
    a2[j] = fmaxf(acc, 0.f);
  }
  float o = BO[0];
  #pragma unroll
  for (int i = 0; i < 10; ++i) o += a2[i] * WO[i];
  OUT[n] = o;
}

__global__ void fill_kernel(float* __restrict__ p, float v, int n)
{
  int i = blockIdx.x * blockDim.x + threadIdx.x;
  if (i < n) p[i] = v;
}

// ============================================================
// Host orchestration
// ============================================================
static void launch_gemm(const float* A, const float* W, const float* RES, float* C,
                        int M, int N, int Kd, int ldA, int ldW, int ldC, int transW,
                        hipStream_t s)
{
  dim3 grid((unsigned)((M / 128) * (N / 64)));
  if (transW) {
    if (RES) gemm_wmma_kernel<1,1><<<grid, 256, 0, s>>>(A, W, RES, C, M, N, Kd, ldA, ldW, ldC);
    else     gemm_wmma_kernel<1,0><<<grid, 256, 0, s>>>(A, W, RES, C, M, N, Kd, ldA, ldW, ldC);
  } else {
    if (RES) gemm_wmma_kernel<0,1><<<grid, 256, 0, s>>>(A, W, RES, C, M, N, Kd, ldA, ldW, ldC);
    else     gemm_wmma_kernel<0,0><<<grid, 256, 0, s>>>(A, W, RES, C, M, N, Kd, ldA, ldW, ldC);
  }
}

struct MBlk { const float *ln_w,*proj_up,*conv_w,*conv_b,*q_w,*k_w,*v_w,*ig_w,*ig_b,*fg_w,*fg_b,*onorm_w,*skip,*proj_down; };
struct SBlk { const float *ln1_w,*ln2_w,*conv_w,*conv_b,*i_w,*f_w,*z_w,*o_w,*rec,*bias,*gn_w,*up_w,*down_w; };

extern "C" void kernel_launch(void* const* d_in, const int* in_sizes, int n_in,
                              void* d_out, int out_size, void* d_ws, size_t ws_size,
                              hipStream_t stream)
{
  (void)in_sizes; (void)n_in; (void)out_size; (void)ws_size;
  int idx = 0;
  auto nxt = [&]() -> const float* { return (const float*)d_in[idx++]; };

  const float* x_in = nxt();
  MBlk mb[7]; SBlk sb;
  for (int i = 0; i < 7; ++i) {
    if (i == 1) {
      sb.ln1_w=nxt(); sb.ln2_w=nxt(); sb.conv_w=nxt(); sb.conv_b=nxt();
      sb.i_w=nxt(); sb.f_w=nxt(); sb.z_w=nxt(); sb.o_w=nxt();
      sb.rec=nxt(); sb.bias=nxt(); sb.gn_w=nxt(); sb.up_w=nxt(); sb.down_w=nxt();
    } else {
      MBlk& m = mb[i];
      m.ln_w=nxt(); m.proj_up=nxt(); m.conv_w=nxt(); m.conv_b=nxt();
      m.q_w=nxt(); m.k_w=nxt(); m.v_w=nxt();
      m.ig_w=nxt(); m.ig_b=nxt(); m.fg_w=nxt(); m.fg_b=nxt();
      m.onorm_w=nxt(); m.skip=nxt(); m.proj_down=nxt();
    }
  }
  const float* post_ln_w = nxt();
  const float* d1w = nxt(); const float* d1b = nxt();
  const float* d2w = nxt(); const float* d2b = nxt();
  const float* ow  = nxt(); const float* ob  = nxt();

  // -------- workspace layout (floats) --------
  float* ws = (float*)d_ws;
  size_t o = 0;
  float* X0 = ws + o; o += (size_t)ROWS_ * D_;
  float* X1 = ws + o; o += (size_t)ROWS_ * D_;
  float* LN = ws + o; o += (size_t)ROWS_ * D_;
  float* UPb= ws + o; o += (size_t)ROWS_ * 1024;
  float* XC = ws + o; o += (size_t)ROWS_ * INNER_;
  float* Qb = ws + o; o += (size_t)ROWS_ * INNER_;
  float* Kb = ws + o; o += (size_t)ROWS_ * INNER_;
  float* Vb = ws + o; o += (size_t)ROWS_ * INNER_;
  float* Hb = ws + o; o += (size_t)ROWS_ * INNER_;
  float* IG = ws + o; o += (size_t)B_ * NH_ * S_;
  float* FG = ws + o; o += (size_t)B_ * NH_ * S_;
  float* STh= ws + o; o += (size_t)B_ * 256;
  float* STc= ws + o; o += (size_t)B_ * 256;
  float* STn= ws + o; o += (size_t)B_ * 256;
  float* STm= ws + o; o += (size_t)B_ * 256;
  float* RAW= ws + o; o += (size_t)B_ * 1024;

  hipMemcpyAsync(X0, x_in, sizeof(float) * (size_t)ROWS_ * D_,
                 hipMemcpyDeviceToDevice, stream);
  float* xcur = X0;
  float* xnxt = X1;

  const int lnGrid  = (ROWS_ + 7) / 8;                 // 8 waves / block
  const int ewGridI = ((ROWS_ * INNER_) + 255) / 256;  // elementwise INNER
  for (int i = 0; i < 7; ++i) {
    if (i == 1) {
      // ---------------- sLSTM block ----------------
      ln_kernel<<<lnGrid, 256, 0, stream>>>(xcur, sb.ln1_w, LN, ROWS_, D_);
      conv_silu_kernel<<<((ROWS_*D_)+255)/256, 256, 0, stream>>>(LN, sb.conv_w, sb.conv_b, XC, D_, D_);
      // headwise 64x64 gate projections (WMMA, W transposed)
      for (int h = 0; h < NH_; ++h) {
        launch_gemm(XC + h*64, sb.i_w + (size_t)h*4096, nullptr, Qb + h*64,
                    ROWS_, 64, 64, D_, 64, D_, 1, stream);
        launch_gemm(XC + h*64, sb.f_w + (size_t)h*4096, nullptr, Kb + h*64,
                    ROWS_, 64, 64, D_, 64, D_, 1, stream);
        launch_gemm(LN + h*64, sb.z_w + (size_t)h*4096, nullptr, Vb + h*64,
                    ROWS_, 64, 64, D_, 64, D_, 1, stream);
        launch_gemm(LN + h*64, sb.o_w + (size_t)h*4096, nullptr, Hb + h*64,
                    ROWS_, 64, 64, D_, 64, D_, 1, stream);
      }
      fill_kernel<<<((B_*256*4)+255)/256, 256, 0, stream>>>(STh, 0.f, B_*256*4);
      // recurrent scan, ys -> UPb (ld D_)
      for (int t = 0; t < S_; ++t) {
        for (int h = 0; h < NH_; ++h)
          launch_gemm(STh + h*64, sb.rec + (size_t)h*16384, nullptr, RAW + h*256,
                      B_, 256, 64, 256, 256, 1024, 0, stream);
        slstm_step_kernel<<<((B_*256)+255)/256, 256, 0, stream>>>(
            RAW, Qb, Kb, Vb, Hb, sb.bias, STh, STc, STn, STm, UPb, t);
      }
      gn_res_kernel<<<((B_*S_*NH_)+7)/8, 256, 0, stream>>>(UPb, sb.gn_w, xcur, xnxt);
      { float* tmp = xcur; xcur = xnxt; xnxt = tmp; }
      // ---------------- FFN ----------------
      ln_kernel<<<lnGrid, 256, 0, stream>>>(xcur, sb.ln2_w, LN, ROWS_, D_);
      launch_gemm(LN, sb.up_w, nullptr, UPb, ROWS_, 2*FFN_, D_, D_, 2*FFN_, 2*FFN_, 0, stream);
      geglu_kernel<<<((ROWS_*FFN_)+255)/256, 256, 0, stream>>>(UPb, Hb);
      launch_gemm(Hb, sb.down_w, xcur, xnxt, ROWS_, D_, FFN_, FFN_, D_, D_, 0, stream);
      { float* tmp = xcur; xcur = xnxt; xnxt = tmp; }
    } else {
      // ---------------- mLSTM block ----------------
      const MBlk& m = mb[i];
      ln_kernel<<<lnGrid, 256, 0, stream>>>(xcur, m.ln_w, LN, ROWS_, D_);
      launch_gemm(LN, m.proj_up, nullptr, UPb, ROWS_, 1024, D_, D_, 1024, 1024, 0, stream);
      conv_silu_kernel<<<((ROWS_*INNER_)+255)/256, 256, 0, stream>>>(
          UPb, m.conv_w, m.conv_b, XC, INNER_, 1024);
      headwise4_kernel<<<ewGridI, 256, 0, stream>>>(XC,  m.q_w, Qb, INNER_);
      headwise4_kernel<<<ewGridI, 256, 0, stream>>>(XC,  m.k_w, Kb, INNER_);
      headwise4_kernel<<<ewGridI, 256, 0, stream>>>(UPb, m.v_w, Vb, 1024);
      gate_kernel<<<(ROWS_+7)/8, 256, 0, stream>>>(
          Qb, Kb, Vb, m.ig_w, m.ig_b, m.fg_w, m.fg_b, IG, FG);
      attn_kernel<<<B_*NH_, 256, 0, stream>>>(
          Qb, Kb, Vb, IG, FG, XC, UPb, m.onorm_w, m.skip, Hb);
      launch_gemm(Hb, m.proj_down, xcur, xnxt, ROWS_, D_, INNER_, INNER_, D_, D_, 0, stream);
      { float* tmp = xcur; xcur = xnxt; xnxt = tmp; }
    }
  }

  head_kernel<<<((B_*24)+255)/256, 256, 0, stream>>>(
      xcur, post_ln_w, d1w, d1b, d2w, d2b, ow, ob, (float*)d_out);
}
